// SEE_40183714021582
// MI455X (gfx1250) — matrix operand order repaired and checked by reading, not compiled, not run on hardware
//
#include <hip/hip_runtime.h>
#include <hip/hip_bf16.h>

typedef __attribute__((ext_vector_type(16))) _Float16 v16h;
typedef __attribute__((ext_vector_type(8)))  _Float16 v8h;
typedef __attribute__((ext_vector_type(8)))  float    v8f;
typedef int v4i __attribute__((__vector_size__(16)));   // matches builtin param

#define B_   64
#define T_   200
#define CIN  80
#define KPW  96            // Cin padded up to multiple of 32 for K-loop
#define C_   1024
#define M_   (B_ * T_)     // 12800 rows
#define BC_  (B_ * C_)     // 65536

// gfx1250 async global->LDS path (guarded: falls back to plain LDS staging)
#if __has_builtin(__builtin_amdgcn_global_load_async_to_lds_b128) && \
    __has_builtin(__builtin_amdgcn_s_wait_asynccnt)
#define ASYNC_LDS 1
#else
#define ASYNC_LDS 0
#endif

// ---------------------------------------------------------------------------
// Pack / convert helpers
// ---------------------------------------------------------------------------

// x [M,80] f32 -> xh [M,96] f16, zero-padded K
__global__ __launch_bounds__(256) void cvt_x_kernel(const float* __restrict__ x,
                                                    _Float16* __restrict__ xh) {
    int i = blockIdx.x * 256 + threadIdx.x;
    if (i >= M_ * KPW) return;
    int m = i / KPW, k = i - m * KPW;
    xh[i] = (k < CIN) ? (_Float16)x[(size_t)m * CIN + k] : (_Float16)0.0f;
}

// w [C, Kin] f32 -> pack [(Kpad/16)][C][16] f16  (B-fragment friendly: lane
// reads 16 contiguous K-halves at fixed N; panel rows contiguous for DMA)
__global__ __launch_bounds__(256) void pack_w_kernel(const float* __restrict__ w,
                                                     int Kin, int Kpad,
                                                     _Float16* __restrict__ out) {
    int i = blockIdx.x * 256 + threadIdx.x;
    if (i >= C_ * Kpad) return;
    int o = i / Kpad, k = i - o * Kpad;
    _Float16 v = (k < Kin) ? (_Float16)w[(size_t)o * Kin + k] : (_Float16)0.0f;
    out[(size_t)(k >> 4) * (C_ * 16) + (size_t)o * 16 + (k & 15)] = v;
}

// eval-BN -> per-channel affine: inv = g*rsqrt(var+eps), bias = b - mean*inv
__global__ __launch_bounds__(256) void bn_prep_kernel(const float* __restrict__ g,
                                                      const float* __restrict__ be,
                                                      const float* __restrict__ mu,
                                                      const float* __restrict__ var,
                                                      float* __restrict__ inv,
                                                      float* __restrict__ bias) {
    int c = blockIdx.x * 256 + threadIdx.x;
    if (c >= C_) return;
    float iv = g[c] * rsqrtf(var[c] + 1e-5f);
    inv[c]  = iv;
    bias[c] = be[c] - mu[c] * iv;
}

__global__ __launch_bounds__(256) void fill_ones_kernel(float* __restrict__ p) {
    int i = blockIdx.x * 256 + threadIdx.x;
    if (i < C_) p[i] = 1.0f;
}

// ---------------------------------------------------------------------------
// WMMA GEMM:  Out[M,1024] = A[M,Kdim](f16) * Bpack, epilogue acc*scale+bias.
//   Bpack layout [Kdim/16][1024][16]  (== W^T tiles, lane-contiguous)
// Block = 256 thr = 8 waves (4M x 2N); block tile 128(M) x 128(N);
// wave tile 32x64 -> 8 WMMA per K-step.
// B panel (2 x 128 x 16 halves = 8KB) double-buffered in LDS, filled with
// gfx1250 async global->LDS DMA (s_wait_asynccnt) when available; A streams
// from global through the 192MB L2 with explicit prefetch of the next K slab.
// ---------------------------------------------------------------------------
__global__ __launch_bounds__(256)
void gemm_wmma_kernel(const _Float16* __restrict__ A, int lda, int Kdim,
                      const _Float16* __restrict__ Bp,
                      const float* __restrict__ scale,
                      const float* __restrict__ bias,
                      float* __restrict__ Out) {
    __shared__ _Float16 bsh[2][2 * 128 * 16];   // 2 buffers x 4096 halves (16KB)

    const int lane  = threadIdx.x & 31;
    const int wave  = threadIdx.x >> 5;
    const int mw    = wave & 3;                 // 0..3  (M wave)
    const int nw    = wave >> 2;                // 0..1  (N wave)
    const int m0    = blockIdx.x * 128 + mw * 32;
    const int nBase = blockIdx.y * 128;         // block's first column
    const int lhalf = lane >> 4;                // 0: lanes 0-15, 1: lanes 16-31
    const int lrow  = lane & 15;

    // ---- cooperative B-panel staging: 4096 halves / 256 thr = 32B/thread ----
    auto stage = [&](int buf, int kk) {
        const int h0  = threadIdx.x * 16;       // 16 halves (32B) per thread
        const int kbl = h0 >> 11;               // which 16-K sub-block (0/1)
        const int rem = h0 & 2047;              // contiguous inside sub-block
        const _Float16* src = Bp + (size_t)((kk >> 4) + kbl) * (C_ * 16) +
                              (size_t)nBase * 16 + rem;
        _Float16* dst = &bsh[buf][h0];
#if ASYNC_LDS
        __builtin_amdgcn_global_load_async_to_lds_b128(
            (v4i*)src, (v4i*)dst, 0, 0);
        __builtin_amdgcn_global_load_async_to_lds_b128(
            (v4i*)(src + 8), (v4i*)(dst + 8), 0, 0);
#else
        *(v8h*)dst       = *(const v8h*)src;
        *(v8h*)(dst + 8) = *(const v8h*)(src + 8);
#endif
    };

    v8f acc[2][4] = {};

    stage(0, 0);
    int buf = 0;
    for (int kk = 0; kk < Kdim; kk += 32) {
#if ASYNC_LDS
        __builtin_amdgcn_s_wait_asynccnt(0);    // my panel writes done
#endif
        __syncthreads();                        // panel visible to all waves
        if (kk + 32 < Kdim) stage(buf ^ 1, kk + 32);   // overlap next DMA

        // A fragment (contiguous K order, same as B): lane holds row M=lane%16,
        // K halves {0..15} (lanes 0-15) / {16..31} (lanes 16-31)
        v16h a[2], b[4];
#pragma unroll
        for (int mi = 0; mi < 2; ++mi) {
            const _Float16* pa =
                A + (size_t)(m0 + mi * 16 + lrow) * lda + kk + lhalf * 16;
            v8h lo = *(const v8h*)pa;
            v8h hi = *(const v8h*)(pa + 8);
            a[mi] = __builtin_shufflevector(lo, hi, 0, 1, 2, 3, 4, 5, 6, 7,
                                            8, 9, 10, 11, 12, 13, 14, 15);
        }
        // B fragment from LDS: lanes 0-15 hold K-block 0, lanes 16-31 block 1,
        // N = lane%16 -> one contiguous 32B LDS read per fragment
        {
            const _Float16* bp0 =
                &bsh[buf][lhalf * 2048 + (nw * 64 + lrow) * 16];
#pragma unroll
            for (int ni = 0; ni < 4; ++ni)
                b[ni] = *(const v16h*)(bp0 + ni * 256);
        }
        if (kk + 32 < Kdim)  // speculative prefetch of next A slab
            __builtin_prefetch(A + (size_t)(m0 + lrow) * lda + kk + 32, 0, 1);

#pragma unroll
        for (int mi = 0; mi < 2; ++mi)
#pragma unroll
            for (int ni = 0; ni < 4; ++ni)
                acc[mi][ni] = __builtin_amdgcn_wmma_f32_16x16x32_f16(
                    false, a[mi], false, b[ni], (short)0, acc[mi][ni],
                    false, false);
        buf ^= 1;
    }

    // C/D layout: VGPR r -> M = r (lanes 0-15) / r+8 (lanes 16-31), N = lane%16
#pragma unroll
    for (int ni = 0; ni < 4; ++ni) {
        const int col = nBase + nw * 64 + ni * 16 + lrow;
        const float s  = scale[col];
        const float bb = bias[col];
#pragma unroll
        for (int mi = 0; mi < 2; ++mi)
#pragma unroll
            for (int r = 0; r < 8; ++r) {
                const int row = m0 + mi * 16 + r + lhalf * 8;
                Out[(size_t)row * C_ + col] = acc[mi][ni][r] * s + bb;
            }
    }
}

// ---------------------------------------------------------------------------
// Fused depthwise-conv(K=7,same) + BN1 + LIF scan #1.
// One thread per (b,c); register ring buffer for the 7-tap window; coalesced
// over c. Emits spikes as f16 (exactly 0/1) in [T,B,C] for the WMMA GEMM.
// LIF (tau=2, reset=0): v = 0.5*v + x ; s = (v>=1) ; v = s?0:v
// ---------------------------------------------------------------------------
__global__ __launch_bounds__(256)
void lif1_dw_bn_kernel(const float* __restrict__ u0,   // [B,T,C] rows b*T+t
                       const float* __restrict__ dw_w, // [C,7]
                       const float* __restrict__ dw_b,
                       const float* __restrict__ inv1,
                       const float* __restrict__ bias1,
                       _Float16* __restrict__ s1h) {   // [T,B,C]
    const int tid = blockIdx.x * 256 + threadIdx.x;    // 0..BC_-1
    const int b = tid >> 10;
    const int c = tid & (C_ - 1);
    const float k0 = dw_w[c * 7 + 0], k1 = dw_w[c * 7 + 1], k2 = dw_w[c * 7 + 2],
                k3 = dw_w[c * 7 + 3], k4 = dw_w[c * 7 + 4], k5 = dw_w[c * 7 + 5],
                k6 = dw_w[c * 7 + 6];
    const float db = dw_b[c], iv = inv1[c], bs = bias1[c];
    const float* up = u0 + (size_t)b * T_ * C_ + c;
    float w0 = 0.f, w1 = 0.f, w2 = 0.f;                 // t-3..t-1 (pad)
    float w3 = up[0], w4 = up[(size_t)C_], w5 = up[(size_t)2 * C_],
          w6 = up[(size_t)3 * C_];
    float v = 0.0f;
    _Float16* sp = s1h + (size_t)b * C_ + c;
    for (int t = 0; t < T_; ++t) {
        float y = k0 * w0 + k1 * w1 + k2 * w2 + k3 * w3 + k4 * w4 + k5 * w5 +
                  k6 * w6 + db;
        y = y * iv + bs;                                // BN1 affine
        v = 0.5f * v + y;                               // LIF integrate
        const bool s = (v >= 1.0f);
        v = s ? 0.0f : v;                               // hard reset
        sp[(size_t)t * BC_] = s ? (_Float16)1.0f : (_Float16)0.0f;
        w0 = w1; w1 = w2; w2 = w3; w3 = w4; w4 = w5; w5 = w6;
        const int tn = t + 4;
        w6 = (tn < T_) ? up[(size_t)tn * C_] : 0.0f;
    }
}

// ---------------------------------------------------------------------------
// LIF scan #2 + residual: out[t,b,c] = spike2 + spike1
// (u2 already carries the BN2 affine from the GEMM epilogue)
// ---------------------------------------------------------------------------
__global__ __launch_bounds__(256)
void lif2_res_kernel(const float* __restrict__ u2,     // [T,B,C]
                     const _Float16* __restrict__ s1h, // [T,B,C]
                     float* __restrict__ out) {        // [T,B,C]
    const int tid = blockIdx.x * 256 + threadIdx.x;    // 0..BC_-1
    float v = 0.0f;
    for (int t = 0; t < T_; ++t) {
        const size_t idx = (size_t)t * BC_ + tid;
        v = 0.5f * v + u2[idx];
        const bool s = (v >= 1.0f);
        v = s ? 0.0f : v;
        out[idx] = (s ? 1.0f : 0.0f) + (float)s1h[idx];
    }
}

// ---------------------------------------------------------------------------
// Launch
// ---------------------------------------------------------------------------
extern "C" void kernel_launch(void* const* d_in, const int* in_sizes, int n_in,
                              void* d_out, int out_size, void* d_ws, size_t ws_size,
                              hipStream_t stream) {
    const float* x       = (const float*)d_in[0];   // [B,T,Cin]
    const float* pw_w    = (const float*)d_in[1];   // [C,Cin]
    const float* pw_b    = (const float*)d_in[2];   // [C]
    const float* dw_w    = (const float*)d_in[3];   // [C,7]
    const float* dw_b    = (const float*)d_in[4];   // [C]
    const float* bn1_g   = (const float*)d_in[5];
    const float* bn1_b   = (const float*)d_in[6];
    const float* bn1_m   = (const float*)d_in[7];
    const float* bn1_v   = (const float*)d_in[8];
    const float* lin_w   = (const float*)d_in[9];   // [C,C]
    const float* bn2_g   = (const float*)d_in[10];
    const float* bn2_b   = (const float*)d_in[11];
    const float* bn2_m   = (const float*)d_in[12];
    const float* bn2_v   = (const float*)d_in[13];
    float* out = (float*)d_out;

    // Workspace carve (u2 aliases u0 — u0 is dead after the LIF1 kernel)
    char* p = (char*)d_ws;
    auto carve = [&](size_t bytes) {
        void* r = (void*)p;
        p += (bytes + 255) & ~(size_t)255;
        return r;
    };
    _Float16* xh   = (_Float16*)carve((size_t)M_ * KPW * 2);   // 2.46 MB
    _Float16* pwp  = (_Float16*)carve((size_t)KPW * C_ * 2);   // 0.20 MB
    _Float16* lwp  = (_Float16*)carve((size_t)C_ * C_ * 2);    // 2.10 MB
    float*    ones = (float*)carve(C_ * 4);
    float*    inv1 = (float*)carve(C_ * 4);
    float*    bia1 = (float*)carve(C_ * 4);
    float*    inv2 = (float*)carve(C_ * 4);
    float*    bia2 = (float*)carve(C_ * 4);
    _Float16* s1h  = (_Float16*)carve((size_t)M_ * C_ * 2);    // 26.2 MB
    float*    u0   = (float*)carve((size_t)M_ * C_ * 4);       // 52.4 MB
    float*    u2   = u0;                                       // alias

    // 1) pack inputs to fp16 / fragment-friendly layouts, fold BN params
    cvt_x_kernel<<<(M_ * KPW + 255) / 256, 256, 0, stream>>>(x, xh);
    pack_w_kernel<<<(C_ * KPW + 255) / 256, 256, 0, stream>>>(pw_w, CIN, KPW, pwp);
    pack_w_kernel<<<(C_ * C_ + 255) / 256, 256, 0, stream>>>(lin_w, C_, C_, lwp);
    fill_ones_kernel<<<(C_ + 255) / 256, 256, 0, stream>>>(ones);
    bn_prep_kernel<<<(C_ + 255) / 256, 256, 0, stream>>>(bn1_g, bn1_b, bn1_m, bn1_v, inv1, bia1);
    bn_prep_kernel<<<(C_ + 255) / 256, 256, 0, stream>>>(bn2_g, bn2_b, bn2_m, bn2_v, inv2, bia2);

    // 2) pointwise conv as WMMA GEMM  [12800,96]x[96,1024] (+pw bias)
    dim3 ggrid(M_ / 128, C_ / 128);
    gemm_wmma_kernel<<<ggrid, 256, 0, stream>>>(xh, KPW, KPW, pwp,
                                                ones, pw_b, u0);

    // 3) fused depthwise conv + BN1 + LIF1 -> fp16 spikes [T,B,C]
    lif1_dw_bn_kernel<<<BC_ / 256, 256, 0, stream>>>(u0, dw_w, dw_b,
                                                     inv1, bia1, s1h);

    // 4) linear layer as WMMA GEMM  [12800,1024]x[1024,1024] (+BN2 affine)
    gemm_wmma_kernel<<<ggrid, 256, 0, stream>>>(s1h, C_, C_, lwp,
                                                inv2, bia2, u2);

    // 5) LIF2 + residual add -> output [T,B,C]
    lif2_res_kernel<<<BC_ / 256, 256, 0, stream>>>(u2, s1h, out);
}
